// CausalAnomalyDetector_28415503630423
// MI455X (gfx1250) — matrix operand order repaired and checked by reading, not compiled, not run on hardware
//
#include <hip/hip_runtime.h>

typedef _Float16 v16h __attribute__((ext_vector_type(16)));
typedef _Float16 v8h  __attribute__((ext_vector_type(8)));
typedef float    v8f  __attribute__((ext_vector_type(8)));
typedef unsigned int u32x4 __attribute__((ext_vector_type(4)));
typedef int          i32x4 __attribute__((ext_vector_type(4)));
typedef int          i32x8 __attribute__((ext_vector_type(8)));

#define FN 6        // factors / nodes
#define HD 32       // hidden dim
#define SPW 64      // samples per workgroup
#define NTH 256     // threads per block (8 wave32)
#define NROWS (SPW*FN)        // 384 (sample,node) rows per WG
#define NPAIR (SPW*FN*FN)     // 2304 pair rows per WG
#define MT    (NPAIR/16)      // 144 edge M-tiles per WG
#define RT    (NROWS/16)      // 24 dynamics/scorer M-tiles per WG
#define NWAVE (NTH/32)

// Sum across each 16-lane row (halves stay independent) via DPP row rotates —
// VALU-only, replaces the ds_bpermute butterfly.
static __device__ __forceinline__ float rowsum16(float v) {
    int x = __float_as_int(v);
    v += __int_as_float(__builtin_amdgcn_update_dpp(x, x, 0x121, 0xF, 0xF, false)); // row_ror:1
    x = __float_as_int(v);
    v += __int_as_float(__builtin_amdgcn_update_dpp(x, x, 0x122, 0xF, 0xF, false)); // row_ror:2
    x = __float_as_int(v);
    v += __int_as_float(__builtin_amdgcn_update_dpp(x, x, 0x124, 0xF, 0xF, false)); // row_ror:4
    x = __float_as_int(v);
    v += __int_as_float(__builtin_amdgcn_update_dpp(x, x, 0x128, 0xF, 0xF, false)); // row_ror:8
    return v;
}

// Build an f16 B fragment (K x N, 16x16x32 layout) from a row-major f32 weight,
// zero-padding K >= kvalid and N >= nvalid.
static __device__ __forceinline__ v16h make_bfrag(const float* __restrict__ W, int nstride,
                                                  int n, int nvalid, int kbase, int kvalid) {
    v16h b;
    #pragma unroll
    for (int e = 0; e < 16; ++e) {
        const int k = (kbase & ~31) + ((kbase & 31) >> 1) + (e < 8 ? e : e + 8);
        const float w = (k < kvalid && n < nvalid) ? W[k * nstride + n] : 0.0f;
        b[e] = (_Float16)w;
    }
    return b;
}

// Load an f16 A fragment (16x32) from an LDS row base (already offset by halfSel*8 halfs).
static __device__ __forceinline__ v16h load_afrag(const _Float16* rowbase) {
    const v8h* p = (const v8h*)rowbase;
    v8h lo = p[0], hi = p[2];
    return __builtin_shufflevector(lo, hi, 0,1,2,3,4,5,6,7,8,9,10,11,12,13,14,15);
}

#define WMMA16(A, B, C) __builtin_amdgcn_wmma_f32_16x16x32_f16(false, (A), false, (B), (short)0, (C), false, false)

__global__ __launch_bounds__(NTH) void causal_fused_kernel(
    const float* __restrict__ fac_g,
    const float* __restrict__ Wn,  const float* __restrict__ bn,
    const float* __restrict__ We1, const float* __restrict__ be1,
    const float* __restrict__ We2, const float* __restrict__ be2,
    const float* __restrict__ Wd1, const float* __restrict__ bd1,
    const float* __restrict__ Wd2, const float* __restrict__ bd2,
    const float* __restrict__ Wd3, const float* __restrict__ bd3,
    const float* __restrict__ Ws1, const float* __restrict__ bs1,
    const float* __restrict__ Ws2, const float* __restrict__ bs2,
    const float* __restrict__ Ws3, const float* __restrict__ bs3,
    float* __restrict__ out_adj, float* __restrict__ out_pred,
    float* __restrict__ out_scores)
{
    __shared__ __align__(16) float    fac_l[SPW * FN * FN];   // 9216 B
    __shared__ __align__(16) _Float16 nf_l[NROWS * HD];       // 24576 B; reused as f16 staging in phase 3
    __shared__ __align__(16) float    adj_l[SPW * FN * FN];   // 9216 B
    __shared__ float Wn_l[FN * HD], bn_l[HD];
    __shared__ float score_acc[SPW];

    const int tid     = threadIdx.x;
    const int sbase   = blockIdx.x * SPW;
    const int lane    = tid & 31;
    const int wave    = tid >> 5;
    const int col     = lane & 15;
    const int halfSel = lane >> 4;

    // ---------------- phase 0: stage factors (TDM) + node-encoder weights ---
#if defined(__gfx1250__) && __has_builtin(__builtin_amdgcn_tensor_load_to_lds)
    if (wave == 0) {
        // 1-D TDM descriptor: 2304 contiguous f32 -> fac_l
        const unsigned long long ga =
            (unsigned long long)(const void*)(fac_g + (size_t)sbase * (FN * FN));
        const unsigned int lds_off = (unsigned int)(unsigned long long)(const void*)&fac_l[0];
        u32x4 g0;
        g0[0] = 1u;                                      // count=1, is_restore=0, gather=0
        g0[1] = lds_off;                                 // lds_addr
        g0[2] = (unsigned int)ga;                        // global_addr[31:0]
        g0[3] = (unsigned int)(ga >> 32) | 0x80000000u;  // global_addr[56:32] | type=2
        i32x8 g1;
        g1[0] = (int)(2u << 16);                         // data_size = 4B
        g1[1] = (int)((NPAIR & 0xFFFFu) << 16);          // tensor_dim0 lo16 (bits 63:48)
        g1[2] = (int)(1u << 16);                         // tensor_dim0 hi16=0 ; tensor_dim1 = 1
        g1[3] = (int)((unsigned)NPAIR << 16);            // tensor_dim1 hi=0 ; tile_dim0 = 2304
        g1[4] = 1;                                       // tile_dim1 = 1, tile_dim2 = 0
        g1[5] = NPAIR;                                   // tensor_dim0_stride lo32
        g1[6] = 0;
        g1[7] = 0;
        i32x4 gz; gz[0] = gz[1] = gz[2] = gz[3] = 0;
#if __clang_major__ >= 23
        i32x8 gz8; gz8[0]=gz8[1]=gz8[2]=gz8[3]=gz8[4]=gz8[5]=gz8[6]=gz8[7]=0;
        __builtin_amdgcn_tensor_load_to_lds(g0, g1, gz, gz, gz8, 0);
#else
        __builtin_amdgcn_tensor_load_to_lds(g0, g1, gz, gz, 0);
#endif
        __builtin_amdgcn_s_wait_tensorcnt(0);
    }
#else
    for (int k = tid; k < SPW * FN * FN; k += NTH)
        fac_l[k] = fac_g[(size_t)sbase * (FN * FN) + k];
#endif
    for (int k = tid; k < FN * HD; k += NTH) Wn_l[k] = Wn[k];
    if (tid < HD) bn_l[tid] = bn[tid];
    if (tid < SPW) score_acc[tid] = 0.0f;
    __syncthreads();

    // ---------------- phase 1: node encoder nf = fac @ Wn + bn (f16 to LDS) -
    for (int t = tid; t < NROWS; t += NTH) {
        const int s = t / FN, i = t - s * FN;
        const float* fr = &fac_l[s * (FN * FN) + i * FN];
        #pragma unroll
        for (int h = 0; h < HD; ++h) {
            float a = bn_l[h];
            #pragma unroll
            for (int k = 0; k < FN; ++k) a += fr[k] * Wn_l[k * HD + h];
            nf_l[t * HD + h] = (_Float16)a;
        }
    }
    __syncthreads();

    // ---------------- phase 2: edge predictor via WMMA ----------------------
    // (B*36) x 64 @ 64 x 32, f16 in / f32 acc; then ReLU . We2, sigmoid, diag=0
    v16h bfr[2][2];
    #pragma unroll
    for (int kt = 0; kt < 2; ++kt)
        #pragma unroll
        for (int nt = 0; nt < 2; ++nt)
            bfr[kt][nt] = make_bfrag(We1, HD, nt * 16 + col, 32, kt * 32 + halfSel * 16, 64);

    const float w2a  = We2[col], w2b = We2[col + 16];
    const float be1a = be1[col], be1b = be1[col + 16];
    const float be2s = be2[0];

    for (int t = wave; t < MT; t += NWAVE) {
        const int r = t * 16 + col;       // this lane's A-row
        const int s = r / 36;
        const int p = r - s * 36;
        const int i = p / 6;
        const int j = p - i * 6;

        // A: K 0..31 = nf[s][i], K 32..63 = nf[s][j]
        v16h a0 = load_afrag(&nf_l[(s * FN + i) * HD + halfSel * 8]);
        v16h a1 = load_afrag(&nf_l[(s * FN + j) * HD + halfSel * 8]);

        v8f c0, c1;
        #pragma unroll
        for (int g = 0; g < 8; ++g) { c0[g] = be1a; c1[g] = be1b; }
        c0 = WMMA16(a0, bfr[0][0], c0);
        c0 = WMMA16(a1, bfr[1][0], c0);
        c1 = WMMA16(a0, bfr[0][1], c1);
        c1 = WMMA16(a1, bfr[1][1], c1);

        v8f part;
        #pragma unroll
        for (int g = 0; g < 8; ++g) {
            const float e0 = c0[g] > 0.0f ? c0[g] : 0.0f;
            const float e1 = c1[g] > 0.0f ? c1[g] : 0.0f;
            part[g] = e0 * w2a + e1 * w2b;
        }
        #pragma unroll
        for (int g = 0; g < 8; ++g) part[g] = rowsum16(part[g]);

        if (col < 8) {
            const int m  = halfSel * 8 + col;
            const int rw = t * 16 + m;
            const int sw = rw / 36;
            const int pw = rw - sw * 36;
            const int iw = pw / 6;
            const int jw = pw - iw * 6;
            const float logit = part[col] + be2s;
            const float av = (iw == jw) ? 0.0f : 1.0f / (1.0f + __expf(-logit));
            adj_l[sw * 36 + pw] = av;
            out_adj[(size_t)(sbase + sw) * 36 + pw] = av;
        }
    }
    __syncthreads();

    // ---------------- phase 3a: structured = einsum(adj, fac), staged f16 ---
    // nf_l is dead; reuse it as [384][32] zero-padded activation staging.
    for (int t = tid; t < NROWS; t += NTH) {
        const int s = t / FN, n = t - s * FN;
        float fa[FN];
        #pragma unroll
        for (int q = 0; q < FN; ++q) fa[q] = fac_l[s * 36 + n * FN + q];
        v8h row;
        #pragma unroll
        for (int i2 = 0; i2 < FN; ++i2) {
            float a = 0.0f;
            #pragma unroll
            for (int j2 = 0; j2 < FN; ++j2) a += adj_l[s * 36 + i2 * FN + j2] * fa[j2];
            row[i2] = (_Float16)a;
        }
        row[6] = (_Float16)0.0f; row[7] = (_Float16)0.0f;
        v8h z;
        #pragma unroll
        for (int e = 0; e < 8; ++e) z[e] = (_Float16)0.0f;
        *(v8h*)&nf_l[t * HD +  0] = row;
        *(v8h*)&nf_l[t * HD +  8] = z;
        *(v8h*)&nf_l[t * HD + 16] = z;
        *(v8h*)&nf_l[t * HD + 24] = z;
    }
    __syncthreads();

    // ---------------- phase 3b: dynamics + scorer as a per-wave WMMA chain --
    // Each wave owns 16-row tiles; activations staged in-place in nf_l (LDS ops
    // from one wave are in-order, so store->load staging needs no barrier).
    const v16h Bd1n0 = make_bfrag(Wd1, HD, col,      32, halfSel * 16,  FN);
    const v16h Bd1n1 = make_bfrag(Wd1, HD, col + 16, 32, halfSel * 16,  FN);
    const v16h Bd2n0 = make_bfrag(Wd2, HD, col,      32, halfSel * 16,  HD);
    const v16h Bd2n1 = make_bfrag(Wd2, HD, col + 16, 32, halfSel * 16,  HD);
    const v16h Bd3f  = make_bfrag(Wd3, FN, col,      FN, halfSel * 16,  HD);
    const v16h Bs1n0 = make_bfrag(Ws1, 32, col,      32, halfSel * 16,  18);
    const v16h Bs1n1 = make_bfrag(Ws1, 32, col + 16, 32, halfSel * 16,  18);
    const v16h Bs2f  = make_bfrag(Ws2, 16, col,      16, halfSel * 16,  32);
    const float bd1a = bd1[col], bd1b = bd1[col + 16];
    const float bd2a = bd2[col], bd2b = bd2[col + 16];
    const float bd3c = (col < FN) ? bd3[col] : 0.0f;
    const float bs1a = bs1[col], bs1b = bs1[col + 16];
    const float bs2c = bs2[col];
    const float w3c  = Ws3[col];
    const float bs3s = bs3[0];

    for (int t3 = wave; t3 < RT; t3 += NWAVE) {
        const int r0 = t3 * 16;
        const int mbase = r0 + 8 * halfSel;                 // D-layout row base for this lane
        const _Float16* arow = &nf_l[(r0 + col) * HD + halfSel * 8];

        v8f d0, d1;
        // h1 = relu(st @ Wd1 + bd1)
        v16h a = load_afrag(arow);
        #pragma unroll
        for (int g = 0; g < 8; ++g) { d0[g] = bd1a; d1[g] = bd1b; }
        d0 = WMMA16(a, Bd1n0, d0);
        d1 = WMMA16(a, Bd1n1, d1);
        #pragma unroll
        for (int g = 0; g < 8; ++g) {
            nf_l[(mbase + g) * HD + col]      = (_Float16)(d0[g] > 0.0f ? d0[g] : 0.0f);
            nf_l[(mbase + g) * HD + col + 16] = (_Float16)(d1[g] > 0.0f ? d1[g] : 0.0f);
        }
        // h2 = relu(h1 @ Wd2 + bd2)
        a = load_afrag(arow);
        #pragma unroll
        for (int g = 0; g < 8; ++g) { d0[g] = bd2a; d1[g] = bd2b; }
        d0 = WMMA16(a, Bd2n0, d0);
        d1 = WMMA16(a, Bd2n1, d1);
        #pragma unroll
        for (int g = 0; g < 8; ++g) {
            nf_l[(mbase + g) * HD + col]      = (_Float16)(d0[g] > 0.0f ? d0[g] : 0.0f);
            nf_l[(mbase + g) * HD + col + 16] = (_Float16)(d1[g] > 0.0f ? d1[g] : 0.0f);
        }
        // pred = h2 @ Wd3 + bd3  (N padded 6->16)
        a = load_afrag(arow);
        v8f dp;
        #pragma unroll
        for (int g = 0; g < 8; ++g) dp[g] = bd3c;
        dp = WMMA16(a, Bd3f, dp);
        // write pred; stage si = [fa, pred, |fa-pred|] (K padded 18->32)
        if (col < FN) {
            #pragma unroll
            for (int g = 0; g < 8; ++g) {
                const int rr = mbase + g;
                const int s  = rr / FN, n = rr - s * FN;
                const float pv = dp[g];
                out_pred[(size_t)(sbase + s) * 36 + n * FN + col] = pv;
                const float fv = fac_l[s * 36 + n * FN + col];
                nf_l[rr * HD + col]          = (_Float16)fv;
                nf_l[rr * HD + FN + col]     = (_Float16)pv;
                nf_l[rr * HD + 2 * FN + col] = (_Float16)fabsf(fv - pv);
            }
        } else {
            #pragma unroll
            for (int g = 0; g < 8; ++g) {
                const int rr = mbase + g;
                nf_l[rr * HD + 12 + col] = (_Float16)0.0f;                 // cols 18..27
                if (col < 10) nf_l[rr * HD + 22 + col] = (_Float16)0.0f;   // cols 28..31
            }
        }
        // s1 = relu(si @ Ws1 + bs1)
        a = load_afrag(arow);
        #pragma unroll
        for (int g = 0; g < 8; ++g) { d0[g] = bs1a; d1[g] = bs1b; }
        d0 = WMMA16(a, Bs1n0, d0);
        d1 = WMMA16(a, Bs1n1, d1);
        #pragma unroll
        for (int g = 0; g < 8; ++g) {
            nf_l[(mbase + g) * HD + col]      = (_Float16)(d0[g] > 0.0f ? d0[g] : 0.0f);
            nf_l[(mbase + g) * HD + col + 16] = (_Float16)(d1[g] > 0.0f ? d1[g] : 0.0f);
        }
        // s2 = relu(s1 @ Ws2 + bs2); s3 = sigmoid(s2 . Ws3 + bs3)
        a = load_afrag(arow);
        v8f ds2;
        #pragma unroll
        for (int g = 0; g < 8; ++g) ds2[g] = bs2c;
        ds2 = WMMA16(a, Bs2f, ds2);
        v8f pv8;
        #pragma unroll
        for (int g = 0; g < 8; ++g) {
            const float e = ds2[g] > 0.0f ? ds2[g] : 0.0f;
            pv8[g] = e * w3c;
        }
        #pragma unroll
        for (int g = 0; g < 8; ++g) pv8[g] = rowsum16(pv8[g]);
        if (col < 8) {
            const int rr = mbase + col;
            const int s  = rr / FN;
            const float sc = 1.0f / (1.0f + __expf(-(pv8[col] + bs3s)));
            atomicAdd(&score_acc[s], sc);
        }
    }
    __syncthreads();

    if (tid < SPW)
        out_scores[sbase + tid] = score_acc[tid] * (1.0f / (float)FN);
}

extern "C" void kernel_launch(void* const* d_in, const int* in_sizes, int n_in,
                              void* d_out, int out_size, void* d_ws, size_t ws_size,
                              hipStream_t stream) {
    const float* fac = (const float*)d_in[0];
    const float* Wn  = (const float*)d_in[1];  const float* bn  = (const float*)d_in[2];
    const float* We1 = (const float*)d_in[3];  const float* be1 = (const float*)d_in[4];
    const float* We2 = (const float*)d_in[5];  const float* be2 = (const float*)d_in[6];
    const float* Wd1 = (const float*)d_in[7];  const float* bd1 = (const float*)d_in[8];
    const float* Wd2 = (const float*)d_in[9];  const float* bd2 = (const float*)d_in[10];
    const float* Wd3 = (const float*)d_in[11]; const float* bd3 = (const float*)d_in[12];
    const float* Ws1 = (const float*)d_in[13]; const float* bs1 = (const float*)d_in[14];
    const float* Ws2 = (const float*)d_in[15]; const float* bs2 = (const float*)d_in[16];
    const float* Ws3 = (const float*)d_in[17]; const float* bs3 = (const float*)d_in[18];

    const int B = in_sizes[0] / (FN * FN);          // 65536
    float* out_adj    = (float*)d_out;              // [B,6,6]
    float* out_pred   = out_adj  + (size_t)B * 36;  // [B,6,6]
    float* out_scores = out_pred + (size_t)B * 36;  // [B]

    const int blocks = B / SPW;                      // 1024
    causal_fused_kernel<<<blocks, NTH, 0, stream>>>(
        fac, Wn, bn, We1, be1, We2, be2,
        Wd1, bd1, Wd2, bd2, Wd3, bd3,
        Ws1, bs1, Ws2, bs2, Ws3, bs3,
        out_adj, out_pred, out_scores);
}
